// CenterPooling_33681133536090
// MI455X (gfx1250) — compile-verified, matrix operand order checked by reading, not run on hardware
//
#include <hip/hip_runtime.h>
#include <hip/hip_bf16.h>

// ---------------- problem constants ----------------
constexpr int Bb  = 4;     // batch
constexpr int CIN = 256;   // input channels
constexpr int C   = 256;   // channels per branch
constexpr int H   = 128;
constexpr int W   = 128;
constexpr int KC  = 8;     // CIN/32 k-chunks of 32 per tap
constexpr int MT_TOTAL = Bb * H * (W / 16);   // 4096 M-tiles of 16 w-positions
constexpr int FRAG = 512;                     // bf16 elems per 32x16 B-fragment

typedef __attribute__((ext_vector_type(16))) __bf16 bf16x16;
typedef __attribute__((ext_vector_type(8)))  __bf16 bf16x8;
typedef __attribute__((ext_vector_type(8)))  float  f32x8;

// Compiler barrier: blocks LICM from hoisting LDS fragment loads out of the
// tile loop (which previously caused a 576-VGPR spill to scratch) and bounds
// per-iteration load batching to one tap's worth of fragments.
#define MEM_FENCE() asm volatile("" ::: "memory")

__device__ __forceinline__ bf16x16 join8(bf16x8 lo, bf16x8 hi) {
  return __builtin_shufflevector(lo, hi, 0,1,2,3,4,5,6,7,8,9,10,11,12,13,14,15);
}
__device__ __forceinline__ f32x8 wmma_bf16(bf16x16 a, bf16x16 b, f32x8 c) {
  // v_wmma_f32_16x16x32_bf16 : D = A(16x32 bf16) x B(32x16 bf16) + C(f32)
  return __builtin_amdgcn_wmma_f32_16x16x32_bf16(false, a, false, b, (short)0, c, false, false);
}

// ---------------- prep: BN fold ----------------
__global__ void k_bn_prep(const float* __restrict__ g, const float* __restrict__ b,
                          const float* __restrict__ m, const float* __restrict__ v,
                          float* __restrict__ sc, float* __restrict__ sh) {
  int c = threadIdx.x;
  float s = g[c] * rsqrtf(v[c] + 1e-5f);
  sc[c] = s;
  sh[c] = b[c] - m[c] * s;
}

// ---------------- prep: NCHW f32 -> NHWC bf16 (LDS-tiled transpose) ----------------
__global__ void k_cvt_x(const float* __restrict__ x, __bf16* __restrict__ xbf) {
  __shared__ float tile[32][33];
  const int w0 = blockIdx.x * 32, c0 = blockIdx.y * 32, bh = blockIdx.z;
  const int b = bh >> 7, h = bh & (H - 1);
  const int tx = threadIdx.x, ty = threadIdx.y;          // 32x8
  const size_t HW = (size_t)H * W;
  #pragma unroll
  for (int i = 0; i < 4; ++i) {
    int c = c0 + ty + i * 8;
    tile[ty + i * 8][tx] = x[((size_t)b * CIN + c) * HW + (size_t)h * W + w0 + tx];
  }
  __syncthreads();
  #pragma unroll
  for (int i = 0; i < 4; ++i) {
    int w = w0 + ty + i * 8;
    xbf[(((size_t)bh * W) + w) * CIN + c0 + tx] = (__bf16)tile[tx][ty + i * 8];
  }
}

// ---------------- prep: pack OIHW f32 weights into WMMA B-fragment order, bf16 ----------------
// dst[((ct*taps + tap)*KCin + kc)*512 + lane*16 + i]
// lane n<16 holds column n with K {kc*32+0..7, kc*32+16..23}; lane n+16 holds K {8..15, 24..31}
__global__ void k_pack_w(const float* __restrict__ src, __bf16* __restrict__ dst,
                         int CT, int taps, int KCin, int Cin) {
  int t = blockIdx.x * blockDim.x + threadIdx.x;
  int total = CT * taps * KCin * FRAG;
  if (t >= total) return;
  int i = t & 15;
  int lane = (t >> 4) & 31;
  int rest = t >> 9;
  int kc = rest % KCin; rest /= KCin;
  int tap = rest % taps; rest /= taps;
  int ct = rest;
  int n = lane & 15, khalf = lane >> 4;
  int kk = (i < 8) ? (khalf * 8 + i) : (16 + khalf * 8 + (i - 8));
  int cin = kc * 32 + kk;
  int cout = ct * 16 + n;
  dst[t] = (__bf16)src[((size_t)cout * Cin + cin) * taps + tap];
}

__global__ void k_zero(float* __restrict__ p, int n) {
  int i = blockIdx.x * blockDim.x + threadIdx.x;
  if (i < n) p[i] = 0.f;
}

// ---------------- kernel 1: dual conv3x3 (up|down, 512 couts) + BN + ReLU, fused axis-max ----------------
// grid (128, 32), block 256. RU[b][h][C] = rowmax(up), CD[b][w][C] = colmax(down) via int atomicMax.
__global__ void k_conv_ud(const __bf16* __restrict__ xbf, const __bf16* __restrict__ wpk,
                          const float* __restrict__ sc_up, const float* __restrict__ sh_up,
                          const float* __restrict__ sc_dn, const float* __restrict__ sh_dn,
                          const __bf16* __restrict__ zpad,
                          float* __restrict__ RU, float* __restrict__ CD) {
  extern __shared__ __attribute__((aligned(64))) char smem[];
  __bf16* lw = (__bf16*)smem;                       // 9*8*512 bf16 = 72 KB
  const int tid = threadIdx.x, ct = blockIdx.y;     // ct 0..31 (0-15 up, 16-31 down)
  const __bf16* wsrc = wpk + (size_t)ct * 9 * KC * FRAG;
  for (int i = tid; i < 9 * KC * FRAG / 8; i += 256)
    ((uint4*)lw)[i] = ((const uint4*)wsrc)[i];
  __syncthreads();
  const int wave = tid >> 5, lane = tid & 31;
  const int nn = lane & 15, khalf = lane >> 4;
  const bool is_up = ct < 16;
  const int cout = (ct & 15) * 16 + nn;
  const float s = is_up ? sc_up[cout] : sc_dn[cout];
  const float t = is_up ? sh_up[cout] : sh_dn[cout];
  for (int mt = blockIdx.x * 8 + wave; mt < MT_TOTAL; mt += 1024) {
    const int wt = mt & 7, row = mt >> 3;
    const int h = row & (H - 1), b = row >> 7;
    const int w0 = wt * 16;
    f32x8 acc = {};
    #pragma unroll
    for (int tap = 0; tap < 9; ++tap) {
      MEM_FENCE();
      const int ky = tap / 3, kx = tap - ky * 3;
      const int hh = h + ky - 1;
      const int ww = w0 + nn + kx - 1;
      const bool valid = ((unsigned)hh < H) && ((unsigned)ww < W);
      // halo -> redirect base pointer to a zeroed pad: unconditional loads,
      // no exec-mask branches, no v_mov zero-fill (no WMMA WAR nops)
      const __bf16* ap = valid
          ? xbf + (((size_t)(b * H + hh) * W + ww) * CIN + khalf * 8)
          : zpad;
      __builtin_prefetch(ap, 0, 1);                  // global_prefetch_b8
      #pragma unroll
      for (int kc = 0; kc < KC; ++kc) {
        bf16x16 a = join8(*(const bf16x8*)(ap + kc * 32),
                          *(const bf16x8*)(ap + kc * 32 + 16));
        bf16x16 bb = *(const bf16x16*)&lw[(tap * KC + kc) * FRAG + lane * 16];
        acc = wmma_bf16(a, bb, acc);
      }
    }
    if (is_up) {  // cummax(rev)∘cummax == broadcast row max -> reduce 16 m's, 1 atomic/lane-pair
      float lm = 0.f;
      #pragma unroll
      for (int r = 0; r < 8; ++r) lm = fmaxf(lm, fmaxf(fmaf(acc[r], s, t), 0.f));
      lm = fmaxf(lm, __shfl_xor(lm, 16, 32));
      if (khalf == 0)
        atomicMax((int*)&RU[((size_t)b * H + h) * C + cout], __float_as_int(lm));
    } else {      // column max over h: one atomic per (w, cout) element (values >= 0)
      #pragma unroll
      for (int r = 0; r < 8; ++r) {
        float v = fmaxf(fmaf(acc[r], s, t), 0.f);
        int w = w0 + r + khalf * 8;
        atomicMax((int*)&CD[((size_t)b * W + w) * C + cout], __float_as_int(v));
      }
    }
  }
}

// ---------------- kernel 2: conv3x3(RU+CD)+BN  +  conv1x1(x)+BN, relu -> relu1 (NHWC bf16) ----------------
__global__ void k_merge(const __bf16* __restrict__ xbf, const __bf16* __restrict__ wpk_p,
                        const __bf16* __restrict__ wpk_c1,
                        const float* __restrict__ RU, const float* __restrict__ CD,
                        const float* __restrict__ sc_p, const float* __restrict__ sh_p,
                        const float* __restrict__ sc_c1, const float* __restrict__ sh_c1,
                        const float* __restrict__ zpadf,
                        __bf16* __restrict__ r1) {
  extern __shared__ __attribute__((aligned(64))) char smem[];
  __bf16* lwp = (__bf16*)smem;                  // 72 KB
  __bf16* lw1 = lwp + 9 * KC * FRAG;            // + 8 KB
  const int tid = threadIdx.x, ct = blockIdx.y; // 0..15
  {
    const __bf16* s0 = wpk_p + (size_t)ct * 9 * KC * FRAG;
    for (int i = tid; i < 9 * KC * FRAG / 8; i += 256) ((uint4*)lwp)[i] = ((const uint4*)s0)[i];
    const __bf16* s1 = wpk_c1 + (size_t)ct * KC * FRAG;
    for (int i = tid; i < KC * FRAG / 8; i += 256) ((uint4*)lw1)[i] = ((const uint4*)s1)[i];
  }
  __syncthreads();
  const int wave = tid >> 5, lane = tid & 31, nn = lane & 15, khalf = lane >> 4;
  const int cout = ct * 16 + nn;
  const float sp = sc_p[cout], tp = sh_p[cout], s1v = sc_c1[cout], t1v = sh_c1[cout];
  for (int mt = blockIdx.x * 8 + wave; mt < MT_TOTAL; mt += 1024) {
    const int wt = mt & 7, row = mt >> 3;
    const int h = row & (H - 1), b = row >> 7;
    const int w0 = wt * 16;
    f32x8 accp = {};
    #pragma unroll
    for (int tap = 0; tap < 9; ++tap) {
      MEM_FENCE();
      const int ky = tap / 3, kx = tap - ky * 3;
      const int hh = h + ky - 1;
      const int ww = w0 + nn + kx - 1;
      const bool valid = ((unsigned)hh < H) && ((unsigned)ww < W);
      const float* rup = valid ? RU + ((size_t)b * H + hh) * C + khalf * 8 : zpadf;
      const float* cdp = valid ? CD + ((size_t)b * W + ww) * C + khalf * 8 : zpadf;
      #pragma unroll
      for (int kc = 0; kc < KC; ++kc) {
        // pooled-sum tensor synthesized on the fly: S = rowmax(up) + colmax(down)
        f32x8 slo = *(const f32x8*)(rup + kc * 32)      + *(const f32x8*)(cdp + kc * 32);
        f32x8 shi = *(const f32x8*)(rup + kc * 32 + 16) + *(const f32x8*)(cdp + kc * 32 + 16);
        bf16x16 a;
        #pragma unroll
        for (int j = 0; j < 8; ++j) { a[j] = (__bf16)slo[j]; a[8 + j] = (__bf16)shi[j]; }
        bf16x16 bb = *(const bf16x16*)&lwp[(tap * KC + kc) * FRAG + lane * 16];
        accp = wmma_bf16(a, bb, accp);
      }
    }
    MEM_FENCE();
    f32x8 acc1 = {};
    const __bf16* ap = xbf + (((size_t)(b * H + h) * W + (w0 + nn)) * CIN + khalf * 8);
    #pragma unroll
    for (int kc = 0; kc < KC; ++kc) {
      bf16x16 a = join8(*(const bf16x8*)(ap + kc * 32), *(const bf16x8*)(ap + kc * 32 + 16));
      bf16x16 bb = *(const bf16x16*)&lw1[kc * FRAG + lane * 16];
      acc1 = wmma_bf16(a, bb, acc1);
    }
    #pragma unroll
    for (int r = 0; r < 8; ++r) {
      float v = fmaxf(fmaf(accp[r], sp, tp) + fmaf(acc1[r], s1v, t1v), 0.f);
      int m = r + khalf * 8;
      r1[(((size_t)(b * H + h) * W) + (w0 + m)) * C + cout] = (__bf16)v;
    }
  }
}

// ---------------- kernel 3: conv3x3(relu1)+BN+ReLU -> out (NCHW f32, LDS-transposed stores) ----------------
__global__ void k_c2(const __bf16* __restrict__ r1, const __bf16* __restrict__ wpk,
                     const float* __restrict__ sc, const float* __restrict__ sh,
                     const __bf16* __restrict__ zpad,
                     float* __restrict__ out) {
  extern __shared__ __attribute__((aligned(64))) char smem[];
  __bf16* lw = (__bf16*)smem;                     // 72 KB weights
  float* lt = (float*)(smem + 9 * KC * FRAG * 2); // 8 waves * 16*17 f32 transpose pads
  const int tid = threadIdx.x, ct = blockIdx.y;
  const __bf16* s0 = wpk + (size_t)ct * 9 * KC * FRAG;
  for (int i = tid; i < 9 * KC * FRAG / 8; i += 256) ((uint4*)lw)[i] = ((const uint4*)s0)[i];
  __syncthreads();
  const int wave = tid >> 5, lane = tid & 31, nn = lane & 15, khalf = lane >> 4;
  const int cout = ct * 16 + nn;
  const float s = sc[cout], t = sh[cout];
  float* myt = lt + wave * 272;
  for (int mt = blockIdx.x * 8 + wave; mt < MT_TOTAL; mt += 1024) {  // uniform 4 iters/wave
    const int wt = mt & 7, row = mt >> 3;
    const int h = row & (H - 1), b = row >> 7;
    const int w0 = wt * 16;
    f32x8 acc = {};
    #pragma unroll
    for (int tap = 0; tap < 9; ++tap) {
      MEM_FENCE();
      const int ky = tap / 3, kx = tap - ky * 3;
      const int hh = h + ky - 1;
      const int ww = w0 + nn + kx - 1;
      const bool valid = ((unsigned)hh < H) && ((unsigned)ww < W);
      const __bf16* ap = valid
          ? r1 + (((size_t)(b * H + hh) * W + ww) * C + khalf * 8)
          : zpad;
      __builtin_prefetch(ap, 0, 1);
      #pragma unroll
      for (int kc = 0; kc < KC; ++kc) {
        bf16x16 a = join8(*(const bf16x8*)(ap + kc * 32),
                          *(const bf16x8*)(ap + kc * 32 + 16));
        bf16x16 bb = *(const bf16x16*)&lw[(tap * KC + kc) * FRAG + lane * 16];
        acc = wmma_bf16(a, bb, acc);
      }
    }
    __syncthreads();                 // WAR vs previous iteration's transpose reads
    #pragma unroll
    for (int r = 0; r < 8; ++r) {
      int m = r + khalf * 8;
      myt[m * 17 + nn] = fmaxf(fmaf(acc[r], s, t), 0.f);
    }
    __syncthreads();
    if (khalf == 0) {                // lane nn owns output channel row -> coalesced float4 stores
      float* ob = out + (((size_t)b * C + cout) * H + h) * W + w0;
      #pragma unroll
      for (int g = 0; g < 4; ++g) {
        float4 v4 = make_float4(myt[(g * 4 + 0) * 17 + nn], myt[(g * 4 + 1) * 17 + nn],
                                myt[(g * 4 + 2) * 17 + nn], myt[(g * 4 + 3) * 17 + nn]);
        ((float4*)ob)[g] = v4;
      }
    }
  }
}

// ---------------- host launcher ----------------
extern "C" void kernel_launch(void* const* d_in, const int* in_sizes, int n_in,
                              void* d_out, int out_size, void* d_ws, size_t ws_size,
                              hipStream_t stream) {
  (void)in_sizes; (void)n_in; (void)out_size; (void)ws_size;
  const float* x = (const float*)d_in[0];
  const float *w_up = (const float*)d_in[1],  *g_up = (const float*)d_in[2],
              *b_up = (const float*)d_in[3],  *m_up = (const float*)d_in[4],  *v_up = (const float*)d_in[5];
  const float *w_dn = (const float*)d_in[6],  *g_dn = (const float*)d_in[7],
              *b_dn = (const float*)d_in[8],  *m_dn = (const float*)d_in[9],  *v_dn = (const float*)d_in[10];
  const float *w_p  = (const float*)d_in[11], *g_p  = (const float*)d_in[12],
              *b_p  = (const float*)d_in[13], *m_p  = (const float*)d_in[14], *v_p  = (const float*)d_in[15];
  const float *w_c1 = (const float*)d_in[16], *g_c1 = (const float*)d_in[17],
              *b_c1 = (const float*)d_in[18], *m_c1 = (const float*)d_in[19], *v_c1 = (const float*)d_in[20];
  const float *w_c2 = (const float*)d_in[21], *g_c2 = (const float*)d_in[22],
              *b_c2 = (const float*)d_in[23], *m_c2 = (const float*)d_in[24], *v_c2 = (const float*)d_in[25];
  float* out = (float*)d_out;

  char* ws = (char*)d_ws;
  size_t off = 0;
  auto take = [&](size_t bytes) -> char* {
    char* p = ws + off;
    off = (off + bytes + 255) & ~(size_t)255;
    return p;
  };
  __bf16* xbf    = (__bf16*)take((size_t)Bb * H * W * CIN * 2);     // 33.5 MB NHWC
  __bf16* r1     = (__bf16*)take((size_t)Bb * H * W * C * 2);       // 33.5 MB NHWC
  __bf16* wpk_ud = (__bf16*)take((size_t)32 * 9 * KC * FRAG * 2);   // up|down packed
  __bf16* wpk_p  = (__bf16*)take((size_t)16 * 9 * KC * FRAG * 2);
  __bf16* wpk_c1 = (__bf16*)take((size_t)16 * 1 * KC * FRAG * 2);
  __bf16* wpk_c2 = (__bf16*)take((size_t)16 * 9 * KC * FRAG * 2);
  // RU, CD, zpadf, zpad allocated contiguously (each 256B-aligned size) and
  // zeroed together by one k_zero launch.
  float*  RU    = (float*)take((size_t)Bb * H * C * 4);             // row maxes  [b][h][c]
  float*  CD    = (float*)take((size_t)Bb * W * C * 4);             // col maxes  [b][w][c]
  float*  zpadf = (float*)take(2048);                               // 512 zero f32 (halo pad)
  __bf16* zpad  = (__bf16*)take(1024);                              // 512 zero bf16 (halo pad)
  float *sc_up = (float*)take(1024), *sh_up = (float*)take(1024);
  float *sc_dn = (float*)take(1024), *sh_dn = (float*)take(1024);
  float *sc_p  = (float*)take(1024), *sh_p  = (float*)take(1024);
  float *sc_c1 = (float*)take(1024), *sh_c1 = (float*)take(1024);
  float *sc_c2 = (float*)take(1024), *sh_c2 = (float*)take(1024);

  // BN fold
  k_bn_prep<<<1, 256, 0, stream>>>(g_up, b_up, m_up, v_up, sc_up, sh_up);
  k_bn_prep<<<1, 256, 0, stream>>>(g_dn, b_dn, m_dn, v_dn, sc_dn, sh_dn);
  k_bn_prep<<<1, 256, 0, stream>>>(g_p,  b_p,  m_p,  v_p,  sc_p,  sh_p);
  k_bn_prep<<<1, 256, 0, stream>>>(g_c1, b_c1, m_c1, v_c1, sc_c1, sh_c1);
  k_bn_prep<<<1, 256, 0, stream>>>(g_c2, b_c2, m_c2, v_c2, sc_c2, sh_c2);
  // activations -> NHWC bf16
  k_cvt_x<<<dim3(W / 32, CIN / 32, Bb * H), dim3(32, 8), 0, stream>>>(x, xbf);
  // weights -> fragment-packed bf16
  k_pack_w<<<2304, 256, 0, stream>>>(w_up, wpk_ud,                        16, 9, KC, CIN);
  k_pack_w<<<2304, 256, 0, stream>>>(w_dn, wpk_ud + 16 * 9 * KC * FRAG,   16, 9, KC, CIN);
  k_pack_w<<<2304, 256, 0, stream>>>(w_p,  wpk_p,                         16, 9, KC, C);
  k_pack_w<<<256,  256, 0, stream>>>(w_c1, wpk_c1,                        16, 1, KC, CIN);
  k_pack_w<<<2304, 256, 0, stream>>>(w_c2, wpk_c2,                        16, 9, KC, C);
  // zero RU + CD + both halo pads (contiguous in ws)
  {
    int nzero = Bb * H * C + Bb * W * C + 512 + 256;  // f32-equivalents
    k_zero<<<(nzero + 255) / 256, 256, 0, stream>>>(RU, nzero);
  }

  // stage 1: both branch convs + BN + ReLU + axis-max reductions (512 couts)
  k_conv_ud<<<dim3(128, 32), 256, 9 * KC * FRAG * 2, stream>>>(
      xbf, wpk_ud, sc_up, sh_up, sc_dn, sh_dn, zpad, RU, CD);
  // stage 2: conv_p(RU⊕CD) + conv_c1(x) + relu -> relu1
  k_merge<<<dim3(128, 16), 256, (9 + 1) * KC * FRAG * 2, stream>>>(
      xbf, wpk_p, wpk_c1, RU, CD, sc_p, sh_p, sc_c1, sh_c1, zpadf, r1);
  // stage 3: conv_c2(relu1) + BN + ReLU -> NCHW f32 output
  k_c2<<<dim3(128, 16), 256, 9 * KC * FRAG * 2 + 8 * 272 * 4, stream>>>(
      r1, wpk_c2, sc_c2, sh_c2, zpad, out);
}
